// FFN_5488968204635
// MI455X (gfx1250) — 8-bit matrix operand order repaired and checked by reading, compiled, not run on hardware
//
#include <hip/hip_runtime.h>
#include <stdint.h>


typedef __attribute__((ext_vector_type(8))) int  v8i;
typedef __attribute__((ext_vector_type(4))) int  v4i;
typedef __attribute__((ext_vector_type(2))) int  v2i;
typedef __attribute__((ext_vector_type(4))) signed char c4;

#define WGSZ 256

__device__ __forceinline__ float warpSum(float v) {
#pragma unroll
  for (int m = 16; m >= 1; m >>= 1) v += __shfl_xor(v, m, 32);
  return v;
}
__device__ __forceinline__ float warpMax(float v) {
#pragma unroll
  for (int m = 16; m >= 1; m >>= 1) v = fmaxf(v, __shfl_xor(v, m, 32));
  return v;
}
__device__ float blockSum(float v, float* smem) {
  __syncthreads();
  const int lane = threadIdx.x & 31, w = threadIdx.x >> 5;
  v = warpSum(v);
  if (lane == 0) smem[w] = v;
  __syncthreads();
  if (w == 0) {
    float t = (threadIdx.x < (blockDim.x >> 5)) ? smem[threadIdx.x] : 0.f;
    t = warpSum(t);
    if (lane == 0) smem[0] = t;
  }
  __syncthreads();
  return smem[0];
}
__device__ float blockMax(float v, float* smem) {
  __syncthreads();
  const int lane = threadIdx.x & 31, w = threadIdx.x >> 5;
  v = warpMax(v);
  if (lane == 0) smem[w] = v;
  __syncthreads();
  if (w == 0) {
    float t = (threadIdx.x < (blockDim.x >> 5)) ? smem[threadIdx.x] : 0.f;
    t = warpMax(t);
    if (lane == 0) smem[0] = t;
  }
  __syncthreads();
  return smem[0];
}

__global__ void __launch_bounds__(WGSZ) absSumKernel(
    const float* __restrict__ w, size_t n, float* __restrict__ partial) {
  __shared__ float smem[32];
  size_t i = (size_t)blockIdx.x * blockDim.x + threadIdx.x;
  const size_t stride = (size_t)gridDim.x * blockDim.x;
  float s = 0.f;
  for (; i < n; i += stride) s += fabsf(w[i]);
  s = blockSum(s, smem);
  if (threadIdx.x == 0) partial[blockIdx.x] = s;
}

__global__ void __launch_bounds__(WGSZ) finalizeScaleKernel(
    const float* __restrict__ partial, int nb, float invN,
    float* __restrict__ outp) {
  __shared__ float smem[32];
  float s = 0.f;
  for (int i = threadIdx.x; i < nb; i += blockDim.x) s += partial[i];
  s = blockSum(s, smem);
  if (threadIdx.x == 0) {
    const float c = fmaxf(s * invN, 1e-5f);
    outp[0] = 1.0f / c;
    outp[1] = c;
  }
}

__global__ void __launch_bounds__(WGSZ) quantWeightKernel(
    const float* __restrict__ w, const float* __restrict__ multPtr,
    int8_t* __restrict__ wq, size_t n4) {
  const size_t i = (size_t)blockIdx.x * blockDim.x + threadIdx.x;
  if (i >= n4) return;
  const float m = *multPtr;
  const float4 v = ((const float4*)w)[i];
  c4 q = {(signed char)(int)rintf(fminf(fmaxf(v.x * m, -1.f), 1.f)),
          (signed char)(int)rintf(fminf(fmaxf(v.y * m, -1.f), 1.f)),
          (signed char)(int)rintf(fminf(fmaxf(v.z * m, -1.f), 1.f)),
          (signed char)(int)rintf(fminf(fmaxf(v.w * m, -1.f), 1.f))};
  *(c4*)(wq + 4 * i) = q;
}

__global__ void __launch_bounds__(WGSZ) rmsActQuantKernel(
    const float* __restrict__ x, const float* __restrict__ nw,
    int8_t* __restrict__ xq, float* __restrict__ deq, int H) {
  __shared__ float smem[32];
  const size_t t = blockIdx.x;
  const float4* xr = (const float4*)(x + t * (size_t)H);
  const float4* wr = (const float4*)nw;
  const int nchunk = H >> 2;
  float4 hl[8];
  int cnt = 0;
  float ss = 0.f;
  for (int c = threadIdx.x; c < nchunk; c += blockDim.x) {
    const float4 v = xr[c], g = wr[c];
    hl[cnt].x = v.x * g.x; hl[cnt].y = v.y * g.y;
    hl[cnt].z = v.z * g.z; hl[cnt].w = v.w * g.w;
    ss += v.x * v.x + v.y * v.y + v.z * v.z + v.w * v.w;
    ++cnt;
  }
  ss = blockSum(ss, smem);
  const float rinv = rsqrtf(ss / (float)H + 1e-6f);
  float amax = 0.f;
  for (int k = 0; k < cnt; ++k) {
    hl[k].x *= rinv; hl[k].y *= rinv; hl[k].z *= rinv; hl[k].w *= rinv;
    amax = fmaxf(amax, fmaxf(fmaxf(fabsf(hl[k].x), fabsf(hl[k].y)),
                             fmaxf(fabsf(hl[k].z), fabsf(hl[k].w))));
  }
  amax = blockMax(amax, smem);
  const float clipped = fmaxf(amax, 1e-5f);
  const float mult = 127.f / clipped;
  c4* out = (c4*)(xq + t * (size_t)H);
  cnt = 0;
  for (int c = threadIdx.x; c < nchunk; c += blockDim.x) {
    const float4 h = hl[cnt++];
    c4 q = {(signed char)(int)rintf(fminf(fmaxf(h.x * mult, -128.f), 127.f)),
            (signed char)(int)rintf(fminf(fmaxf(h.y * mult, -128.f), 127.f)),
            (signed char)(int)rintf(fminf(fmaxf(h.z * mult, -128.f), 127.f)),
            (signed char)(int)rintf(fminf(fmaxf(h.w * mult, -128.f), 127.f))};
    out[c] = q;
  }
  if (threadIdx.x == 0) deq[t] = clipped / 127.f;
}

__global__ void __launch_bounds__(WGSZ) actQuant2Kernel(
    const float* __restrict__ h1, int8_t* __restrict__ hq,
    float* __restrict__ deq, int F) {
  __shared__ float smem[32];
  const size_t t = blockIdx.x;
  const float4* hr = (const float4*)(h1 + t * (size_t)F);
  const int nchunk = F >> 2;
  float4 hl[8];
  int cnt = 0;
  float amax = 0.f;
  for (int c = threadIdx.x; c < nchunk; c += blockDim.x) {
    const float4 v = hr[c];
    hl[cnt++] = v;
    amax = fmaxf(amax, fmaxf(fmaxf(v.x, v.y), fmaxf(v.z, v.w)));
  }
  amax = blockMax(amax, smem);
  const float clipped = fmaxf(amax, 1e-5f);
  const float mult = 127.f / clipped;
  c4* out = (c4*)(hq + t * (size_t)F);
  cnt = 0;
  for (int c = threadIdx.x; c < nchunk; c += blockDim.x) {
    const float4 v = hl[cnt++];
    c4 q = {(signed char)(int)rintf(fminf(v.x * mult, 127.f)),
            (signed char)(int)rintf(fminf(v.y * mult, 127.f)),
            (signed char)(int)rintf(fminf(v.z * mult, 127.f)),
            (signed char)(int)rintf(fminf(v.w * mult, 127.f))};
    out[c] = q;
  }
  if (threadIdx.x == 0) deq[t] = clipped / 127.f;
}

#define TM 128
#define TN 64
#define TK 64
#define LDST 80

template <int MODE>
__global__ void __launch_bounds__(WGSZ) gemmIU8Kernel(
    const int8_t* __restrict__ Aq, const int8_t* __restrict__ Bq,
    const float* __restrict__ rowDeq, const float* __restrict__ wDeqPtr,
    const float* __restrict__ residual, float* __restrict__ out,
    int M, int N, int K) {
  __shared__ int8_t As[TM * LDST];
  __shared__ int8_t Bs[TN * LDST];

  const int tid   = threadIdx.x;
  const int lane  = tid & 31;
  const int wave  = tid >> 5;
  const int waveM = wave & 3;
  const int waveN = wave >> 2;

  const size_t tileM = (size_t)blockIdx.y * TM;
  const size_t tileN = (size_t)blockIdx.x * TN;

  const int ldRow = tid >> 2;
  const int ldCol = (tid & 3) << 4;
  const int8_t* aSrc = Aq + (tileM + ldRow) * (size_t)K + ldCol;
  const int8_t* bSrc = Bq + (tileN + ldRow) * (size_t)K + ldCol;
  const size_t aHalf = (size_t)64 * K;

  v4i aReg0 = *(const v4i*)aSrc;
  v4i aReg1 = *(const v4i*)(aSrc + aHalf);
  v4i bReg  = *(const v4i*)bSrc;

  v8i acc00 = {}, acc01 = {}, acc10 = {}, acc11 = {};

  const int mRow  = lane & 15;
  const int khalf = lane >> 4;

  const int nK = K / TK;
  for (int kt = 0; kt < nK; ++kt) {
    *(v4i*)(As + ldRow * LDST + ldCol) = aReg0;
    *(v4i*)(As + (ldRow + 64) * LDST + ldCol) = aReg1;
    *(v4i*)(Bs + ldRow * LDST + ldCol) = bReg;
    __syncthreads();

    if (kt + 1 < nK) {
      aReg0 = *(const v4i*)(aSrc + (size_t)(kt + 1) * TK);
      aReg1 = *(const v4i*)(aSrc + aHalf + (size_t)(kt + 1) * TK);
      bReg  = *(const v4i*)(bSrc + (size_t)(kt + 1) * TK);
    }
    if (kt + 2 < nK) {
      __builtin_prefetch(aSrc + (size_t)(kt + 2) * TK, 0, 1);
      __builtin_prefetch(aSrc + aHalf + (size_t)(kt + 2) * TK, 0, 1);
      __builtin_prefetch(bSrc + (size_t)(kt + 2) * TK, 0, 1);
    }

    v8i af[2];
#pragma unroll
    for (int am = 0; am < 2; ++am) {
      const int8_t* ab =
          As + (waveM * 32 + am * 16 + mRow) * LDST + (khalf << 4);
      const v2i q0 = *(const v2i*)(ab + 0);
      const v2i q1 = *(const v2i*)(ab + 8);
      const v2i q2 = *(const v2i*)(ab + 32);
      const v2i q3 = *(const v2i*)(ab + 40);
      af[am] = (v8i){q0.x, q0.y, q1.x, q1.y, q2.x, q2.y, q3.x, q3.y};
    }
    v8i bf[2];
#pragma unroll
    for (int bn = 0; bn < 2; ++bn) {
      const int8_t* bb =
          Bs + (waveN * 32 + bn * 16 + mRow) * LDST + (khalf << 4);
      bf[bn].lo = *(const v4i*)(bb);
      bf[bn].hi = *(const v4i*)(bb + 32);
    }

    acc00 = __builtin_amdgcn_wmma_i32_16x16x64_iu8(true, af[0], true, bf[0],
                                                   acc00, false, false);
    acc01 = __builtin_amdgcn_wmma_i32_16x16x64_iu8(true, af[0], true, bf[1],
                                                   acc01, false, false);
    acc10 = __builtin_amdgcn_wmma_i32_16x16x64_iu8(true, af[1], true, bf[0],
                                                   acc10, false, false);
    acc11 = __builtin_amdgcn_wmma_i32_16x16x64_iu8(true, af[1], true, bf[1],
                                                   acc11, false, false);
    __syncthreads();
  }

  const float wdeq = *wDeqPtr;
#pragma unroll
  for (int am = 0; am < 2; ++am) {
#pragma unroll
    for (int bn = 0; bn < 2; ++bn) {
      const v8i acc = am ? (bn ? acc11 : acc10) : (bn ? acc01 : acc00);
      const size_t col = tileN + (size_t)(waveN * 32 + bn * 16 + (lane & 15));
#pragma unroll
      for (int i = 0; i < 8; ++i) {
        const size_t row =
            tileM + (size_t)(waveM * 32 + am * 16 + (lane >> 4) * 8 + i);
        float v = (float)acc[i] * rowDeq[row] * wdeq;
        if (MODE == 0) {
          v = v > 0.f ? v * v : 0.f;
          out[row * (size_t)N + col] = v;
        } else {
          out[row * (size_t)N + col] = v + residual[row * (size_t)N + col];
        }
      }
    }
  }
}

extern "C" void kernel_launch(void* const* d_in, const int* in_sizes, int n_in,
                              void* d_out, int out_size, void* d_ws,
                              size_t ws_size, hipStream_t stream) {
  (void)n_in; (void)out_size; (void)ws_size;
  const float* x      = (const float*)d_in[0];
  const float* norm_w = (const float*)d_in[1];
  const float* w_up   = (const float*)d_in[2];
  const float* w_down = (const float*)d_in[3];
  float* out = (float*)d_out;

  const int H = in_sizes[1];
  const int M = (int)((long long)in_sizes[0] / H);
  const int F = (int)(in_sizes[2] / H);

  uint8_t* ws = (uint8_t*)d_ws;
  size_t off = 0;
  auto alloc = [&](size_t bytes) {
    size_t o = off;
    off += (bytes + 255) & ~(size_t)255;
    return o;
  };
  float*  partial = (float*)(ws + alloc(1024 * sizeof(float)));
  float*  scales  = (float*)(ws + alloc(4 * sizeof(float)));
  float*  xdeq    = (float*)(ws + alloc((size_t)M * sizeof(float)));
  float*  hdeq    = (float*)(ws + alloc((size_t)M * sizeof(float)));
  int8_t* wupq    = (int8_t*)(ws + alloc((size_t)F * H));
  int8_t* wdnq    = (int8_t*)(ws + alloc((size_t)F * H));
  int8_t* xq      = (int8_t*)(ws + alloc((size_t)M * H));
  int8_t* hq      = (int8_t*)(ws + alloc((size_t)M * F));
  float*  h1      = (float*)(ws + alloc((size_t)M * F * sizeof(float)));

  const size_t nW = (size_t)F * H;

  absSumKernel<<<1024, WGSZ, 0, stream>>>(w_up, nW, partial);
  finalizeScaleKernel<<<1, WGSZ, 0, stream>>>(partial, 1024, 1.0f / (float)nW,
                                              scales + 0);
  absSumKernel<<<1024, WGSZ, 0, stream>>>(w_down, nW, partial);
  finalizeScaleKernel<<<1, WGSZ, 0, stream>>>(partial, 1024, 1.0f / (float)nW,
                                              scales + 2);
  const unsigned qBlocks = (unsigned)((nW / 4 + WGSZ - 1) / WGSZ);
  quantWeightKernel<<<qBlocks, WGSZ, 0, stream>>>(w_up, scales + 0, wupq,
                                                  nW / 4);
  quantWeightKernel<<<qBlocks, WGSZ, 0, stream>>>(w_down, scales + 2, wdnq,
                                                  nW / 4);

  rmsActQuantKernel<<<M, WGSZ, 0, stream>>>(x, norm_w, xq, xdeq, H);

  dim3 g1(F / TN, M / TM);
  gemmIU8Kernel<0><<<g1, WGSZ, 0, stream>>>(xq, wupq, xdeq, scales + 1,
                                            nullptr, h1, M, F, H);

  actQuant2Kernel<<<M, WGSZ, 0, stream>>>(h1, hq, hdeq, F);

  dim3 g2(H / TN, M / TM);
  gemmIU8Kernel<1><<<g2, WGSZ, 0, stream>>>(hq, wdnq, hdeq, scales + 3, x, out,
                                            M, H, F);
}
